// PST_11261404250540
// MI455X (gfx1250) — compile-verified, matrix operand order checked by reading, not run on hardware
//
// PatchShift Transformer (PST) block for MI455X / gfx1250 (CDNA5, wave32).
// All GEMMs via v_wmma_f32_16x16x32_bf16; flash-attention with online softmax;
// patch-shift reduced to head-group permutation (D == NH == 8 makes the torch
// .view roll axis the head axis).
#include <hip/hip_runtime.h>
#include <hip/hip_bf16.h>

typedef __attribute__((ext_vector_type(16))) __bf16 v16bf;
typedef __attribute__((ext_vector_type(8)))  float  v8f;

#define B_    2
#define C_    512
#define NH_   8
#define N_    1568          // D*H*W = 8*14*14
#define M_    3136          // B_*N_
#define HID_  2048
#define LDK   40            // 32 + 8 bf16 pad (16B) -> conflict-free LDS rows

// (h%3, w%3) -> temporal (== head) shift, row-major 3x3; (1,1)=0
__device__ __constant__ int c_shift[9] = {-4, 1, 2, -1, 0, 3, -2, -3, 4};

union Frag {
    v16bf v;
    float4 q[2];            // two contiguous 16B runs per lane
    unsigned short u[16];
};

static __device__ __forceinline__ __bf16 f2bf(float x) { return (__bf16)x; }

static __device__ __forceinline__ v8f wmma_bf16(const Frag& a, const Frag& b, v8f c) {
    return __builtin_amdgcn_wmma_f32_16x16x32_bf16(false, a.v, false, b.v,
                                                   (short)0, c, false, false);
}

// ---------------------------------------------------------------- utilities
__global__ void cvt_bf16_kernel(const float* __restrict__ src,
                                __bf16* __restrict__ dst, int n) {
    int i = blockIdx.x * 256 + threadIdx.x;
    if (i < n) dst[i] = f2bf(src[i]);
}

// x (B,C,W,H,D) -> xp (B,N,C), n = d*196 + h*14 + w
__global__ void transpose_in_kernel(const float* __restrict__ x,
                                    float* __restrict__ xp) {
    int idx = blockIdx.x * 256 + threadIdx.x;       // over B*N*C
    if (idx >= B_ * N_ * C_) return;
    int cc = idx & 511;
    int bn = idx >> 9;
    int n = bn % N_, b = bn / N_;
    int d = n / 196, sp = n % 196, hh = sp / 14, ww = sp % 14;
    xp[idx] = x[(((size_t)(b * C_ + cc) * 14 + ww) * 14 + hh) * 8 + d];
}

__global__ void transpose_out_kernel(const float* __restrict__ xp,
                                     float* __restrict__ out) {
    int idx = blockIdx.x * 256 + threadIdx.x;
    if (idx >= B_ * N_ * C_) return;
    int cc = idx & 511;
    int bn = idx >> 9;
    int n = bn % N_, b = bn / N_;
    int d = n / 196, sp = n % 196, hh = sp / 14, ww = sp % 14;
    out[(((size_t)(b * C_ + cc) * 14 + ww) * 14 + hh) * 8 + d] = xp[idx];
}

// ------------------------------------------------- LayerNorm (+fwd shift)
// one wave per row of 512; shiftMode: 0 = none, 1 = forward patch shift
__global__ __launch_bounds__(128) void ln_kernel(const float* __restrict__ X,
                                                 const float* __restrict__ w,
                                                 const float* __restrict__ bia,
                                                 __bf16* __restrict__ out,
                                                 int shiftMode) {
    __shared__ float buf[4][C_];
    int wave = threadIdx.x >> 5, lane = threadIdx.x & 31;
    int row = blockIdx.x * 4 + wave;
    if (row >= M_) return;
    const float* xr = X + (size_t)row * C_;
    float s = 0.f, s2 = 0.f;
#pragma unroll
    for (int i = 0; i < 16; ++i) { float v = xr[lane + i * 32]; s += v; s2 += v * v; }
#pragma unroll
    for (int m = 1; m < 32; m <<= 1) { s += __shfl_xor(s, m, 32); s2 += __shfl_xor(s2, m, 32); }
    float mu = s * (1.f / 512.f);
    float var = s2 * (1.f / 512.f) - mu * mu;
    float rstd = rsqrtf(var + 1e-5f);
#pragma unroll
    for (int i = 0; i < 16; ++i) {
        int cc = lane + i * 32;
        buf[wave][cc] = (xr[cc] - mu) * rstd * w[cc] + bia[cc];
    }
    int sp = row % N_ % 196;
    int s0 = c_shift[((sp / 14) % 3) * 3 + (sp % 14) % 3];
#pragma unroll
    for (int i = 0; i < 16; ++i) {
        int cc = lane + i * 32;
        int head = cc >> 6, ci = cc & 63;
        int src = shiftMode ? ((head - s0) & 7) : head;
        out[(size_t)row * C_ + cc] = f2bf(buf[wave][src * 64 + ci]);
    }
}

// inverse shift: o (B,NH,N,64) bf16 -> (B,N,C) bf16
__global__ void inv_shift_kernel(const __bf16* __restrict__ o,
                                 __bf16* __restrict__ out) {
    int idx = blockIdx.x * 256 + threadIdx.x;
    if (idx >= B_ * N_ * C_) return;
    int cc = idx & 511;
    int bn = idx >> 9;
    int n = bn % N_, b = bn / N_;
    int sp = n % 196;
    int s0 = c_shift[((sp / 14) % 3) * 3 + (sp % 14) % 3];
    int head = cc >> 6, ci = cc & 63;
    int src = (head + s0) & 7;
    out[idx] = o[((size_t)(b * NH_ + src) * N_ + n) * 64 + ci];
}

// --------------------------------------------------------- WMMA bf16 GEMM
// out[g, o] = sum_k A[g,k] * Wt[o,k] + bias[o], BM=64 BN=128 BK=32, 8 waves.
// mode 0: QKV split (->q*0.125,k,v in (B,NH,N,64) bf16)
// mode 1: residual f32: outF = resid + acc
// mode 2: exact GELU -> bf16
__global__ __launch_bounds__(256) void gemm_kernel(
        const __bf16* __restrict__ A, const __bf16* __restrict__ Wt,
        const float* __restrict__ bias, const float* __restrict__ resid,
        float* __restrict__ outF, __bf16* __restrict__ outB,
        __bf16* __restrict__ qO, __bf16* __restrict__ kO, __bf16* __restrict__ vO,
        int K, int Ncols, int mode) {
    __shared__ __align__(16) __bf16 sA[64][LDK];
    __shared__ __align__(16) __bf16 sB[128][LDK];
    int tid = threadIdx.x;
    int m0 = blockIdx.x * 64;
    int n0 = blockIdx.y * 128;
    int wave = tid >> 5, lane = tid & 31;
    int wm = wave >> 1, wn = wave & 1;
    int half = lane >> 4, lidx = lane & 15;
    int lr = tid >> 2;              // 0..63
    int lk = (tid & 3) * 8;         // 0,8,16,24 (bf16)

    v8f acc[4] = {};
    for (int k0 = 0; k0 < K; k0 += 32) {
        *(float4*)&sA[lr][lk]      = *(const float4*)&A[(size_t)(m0 + lr) * K + k0 + lk];
        *(float4*)&sB[lr][lk]      = *(const float4*)&Wt[(size_t)(n0 + lr) * K + k0 + lk];
        *(float4*)&sB[64 + lr][lk] = *(const float4*)&Wt[(size_t)(n0 + 64 + lr) * K + k0 + lk];
        if (k0 + 32 < K) {          // global_prefetch_b8 the next K slab
            __builtin_prefetch(&A[(size_t)(m0 + lr) * K + k0 + 32 + lk], 0, 1);
            __builtin_prefetch(&Wt[(size_t)(n0 + lr) * K + k0 + 32 + lk], 0, 1);
            __builtin_prefetch(&Wt[(size_t)(n0 + 64 + lr) * K + k0 + 32 + lk], 0, 1);
        }
        __syncthreads();
        Frag a;
        const __bf16* ar = &sA[wm * 16 + lidx][0];
        a.q[0] = *(const float4*)(ar + half * 8);
        a.q[1] = *(const float4*)(ar + 16 + half * 8);
#pragma unroll
        for (int nt = 0; nt < 4; ++nt) {
            Frag b;
            const __bf16* br = &sB[wn * 64 + nt * 16 + lidx][0];
            b.q[0] = *(const float4*)(br + half * 8);
            b.q[1] = *(const float4*)(br + 16 + half * 8);
            acc[nt] = wmma_bf16(a, b, acc[nt]);
        }
        __syncthreads();
    }
#pragma unroll
    for (int nt = 0; nt < 4; ++nt)
#pragma unroll
        for (int r = 0; r < 8; ++r) {
            int gcol = n0 + wn * 64 + nt * 16 + lidx;
            int grow = m0 + wm * 16 + half * 8 + r;
            float vv = acc[nt][r] + bias[gcol];
            if (mode == 0) {
                int bb = grow / N_, nn = grow - bb * N_;
                int sec = gcol >> 9, cc = gcol & 511;
                int head = cc >> 6, ci = cc & 63;
                size_t di = ((size_t)(bb * NH_ + head) * N_ + nn) * 64 + ci;
                if (sec == 0)      qO[di] = f2bf(vv * 0.125f);   // * SCALE
                else if (sec == 1) kO[di] = f2bf(vv);
                else               vO[di] = f2bf(vv);
            } else if (mode == 1) {
                size_t di = (size_t)grow * Ncols + gcol;
                outF[di] = resid[di] + vv;
            } else {
                float g = 0.5f * vv * (1.0f + erff(vv * 0.70710678118f));
                outB[(size_t)grow * Ncols + gcol] = f2bf(g);
            }
        }
}

// ----------------------------------------------- flash attention (per head)
// one wave = 16 query rows; 49 col-tiles of 32; online softmax; c = 64.
__global__ __launch_bounds__(128) void attn_kernel(const __bf16* __restrict__ Q,
                                                   const __bf16* __restrict__ Km,
                                                   const __bf16* __restrict__ Vm,
                                                   __bf16* __restrict__ O) {
    __shared__ __align__(16) __bf16 sP[4][16][LDK];     // wave-private P tiles
    int wave = threadIdx.x >> 5, lane = threadIdx.x & 31;
    int half = lane >> 4, lidx = lane & 15;
    int bh = blockIdx.x / 25;
    int tile = (blockIdx.x % 25) * 4 + wave;
    if (tile >= 98) return;
    size_t base = (size_t)bh * N_ * 64;

    Frag aq0, aq1;
    const __bf16* qrow = Q + base + (size_t)(tile * 16 + lidx) * 64;
    aq0.q[0] = *(const float4*)(qrow + half * 8);
    aq0.q[1] = *(const float4*)(qrow + 16 + half * 8);
    aq1.q[0] = *(const float4*)(qrow + 32 + half * 8);
    aq1.q[1] = *(const float4*)(qrow + 48 + half * 8);

    v8f oacc[4] = {};
    float rmax[8], rsum[8];
#pragma unroll
    for (int r = 0; r < 8; ++r) { rmax[r] = -3.0e38f; rsum[r] = 0.f; }

    for (int ct = 0; ct < 49; ++ct) {
        v8f s0 = {}, s1 = {};
        const __bf16* kb = Km + base + (size_t)(ct * 32) * 64;
        {
            Frag b;
            const __bf16* kr = kb + (size_t)lidx * 64;          // cols 0..15
            b.q[0] = *(const float4*)(kr + half * 8);
            b.q[1] = *(const float4*)(kr + 16 + half * 8);
            s0 = wmma_bf16(aq0, b, s0);
            b.q[0] = *(const float4*)(kr + 32 + half * 8);
            b.q[1] = *(const float4*)(kr + 48 + half * 8);
            s0 = wmma_bf16(aq1, b, s0);
            kr = kb + (size_t)(16 + lidx) * 64;                 // cols 16..31
            b.q[0] = *(const float4*)(kr + half * 8);
            b.q[1] = *(const float4*)(kr + 16 + half * 8);
            s1 = wmma_bf16(aq0, b, s1);
            b.q[0] = *(const float4*)(kr + 32 + half * 8);
            b.q[1] = *(const float4*)(kr + 48 + half * 8);
            s1 = wmma_bf16(aq1, b, s1);
        }
#pragma unroll
        for (int r = 0; r < 8; ++r) {                  // row m = half*8 + r
            float tmax = fmaxf(s0[r], s1[r]);
#pragma unroll
            for (int m = 1; m < 16; m <<= 1) tmax = fmaxf(tmax, __shfl_xor(tmax, m, 32));
            float mnew = fmaxf(rmax[r], tmax);
            float alpha = __expf(rmax[r] - mnew);
            rmax[r] = mnew;
            float e0 = __expf(s0[r] - mnew);
            float e1 = __expf(s1[r] - mnew);
            float ts = e0 + e1;
#pragma unroll
            for (int m = 1; m < 16; m <<= 1) ts += __shfl_xor(ts, m, 32);
            rsum[r] = rsum[r] * alpha + ts;
#pragma unroll
            for (int c2 = 0; c2 < 4; ++c2) oacc[c2][r] *= alpha;
            sP[wave][half * 8 + r][lidx]      = f2bf(e0);
            sP[wave][half * 8 + r][16 + lidx] = f2bf(e1);
        }
        Frag ap;                                        // P as 16x32 A-frag
        const __bf16* pr = &sP[wave][lidx][0];
        ap.q[0] = *(const float4*)(pr + half * 8);
        ap.q[1] = *(const float4*)(pr + 16 + half * 8);
        const __bf16* vb = Vm + base + (size_t)(ct * 32) * 64;
#pragma unroll
        for (int c2 = 0; c2 < 4; ++c2) {
            Frag bv;                                    // V is K-major here
#pragma unroll
            for (int e = 0; e < 16; ++e) {
                int kk = (e >> 3) * 16 + half * 8 + (e & 7);
                bv.u[e] = *(const unsigned short*)&vb[(size_t)kk * 64 + c2 * 16 + lidx];
            }
            oacc[c2] = wmma_bf16(ap, bv, oacc[c2]);
        }
    }
#pragma unroll
    for (int c2 = 0; c2 < 4; ++c2)
#pragma unroll
        for (int r = 0; r < 8; ++r) {
            float vv = oacc[c2][r] / rsum[r];
            O[base + (size_t)(tile * 16 + half * 8 + r) * 64 + c2 * 16 + lidx] = f2bf(vv);
        }
}

// ------------------------------------------------------------------- host
extern "C" void kernel_launch(void* const* d_in, const int* in_sizes, int n_in,
                              void* d_out, int out_size, void* d_ws, size_t ws_size,
                              hipStream_t stream) {
    const float* x      = (const float*)d_in[0];
    const float* n1w    = (const float*)d_in[1];
    const float* n1b    = (const float*)d_in[2];
    const float* qkv_w  = (const float*)d_in[3];
    const float* qkv_b  = (const float*)d_in[4];
    const float* proj_w = (const float*)d_in[5];
    const float* proj_b = (const float*)d_in[6];
    const float* n2w    = (const float*)d_in[7];
    const float* n2b    = (const float*)d_in[8];
    const float* fc1_w  = (const float*)d_in[9];
    const float* fc1_b  = (const float*)d_in[10];
    const float* fc2_w  = (const float*)d_in[11];
    const float* fc2_b  = (const float*)d_in[12];

    char* ws = (char*)d_ws;
    size_t off = 0;
    auto take = [&](size_t bytes) -> char* {
        char* p = ws + off;
        off += (bytes + 255) & ~(size_t)255;
        return p;
    };
    float*  xp   = (float*)take((size_t)M_ * C_ * 4);
    float*  y    = (float*)take((size_t)M_ * C_ * 4);
    __bf16* hsh  = (__bf16*)take((size_t)M_ * C_ * 2);
    __bf16* qb   = (__bf16*)take((size_t)M_ * C_ * 2);
    __bf16* kb   = (__bf16*)take((size_t)M_ * C_ * 2);
    __bf16* vb   = (__bf16*)take((size_t)M_ * C_ * 2);
    __bf16* oh   = (__bf16*)take((size_t)M_ * C_ * 2);
    __bf16* osh  = (__bf16*)take((size_t)M_ * C_ * 2);
    __bf16* mbf  = (__bf16*)take((size_t)M_ * C_ * 2);
    __bf16* hid  = (__bf16*)take((size_t)M_ * HID_ * 2);
    __bf16* wqkv = (__bf16*)take((size_t)2 * 1536 * 512 * 2);
    __bf16* wprj = (__bf16*)take((size_t)2 * 512 * 512 * 2);
    __bf16* wfc1 = (__bf16*)take((size_t)2 * 2048 * 512 * 2);
    __bf16* wfc2 = (__bf16*)take((size_t)2 * 512 * 2048 * 2);

    auto cvt = [&](const float* s, __bf16* d, int n) {
        cvt_bf16_kernel<<<(n + 255) / 256, 256, 0, stream>>>(s, d, n);
    };
    cvt(qkv_w, wqkv, 2 * 1536 * 512);
    cvt(proj_w, wprj, 2 * 512 * 512);
    cvt(fc1_w, wfc1, 2 * 2048 * 512);
    cvt(fc2_w, wfc2, 2 * 512 * 2048);

    const int ELEM_BLKS = (B_ * N_ * C_) / 256;   // 6272
    transpose_in_kernel<<<ELEM_BLKS, 256, 0, stream>>>(x, xp);

    for (int d = 0; d < 2; ++d) {
        ln_kernel<<<M_ / 4, 128, 0, stream>>>(xp, n1w + d * C_, n1b + d * C_, hsh, 1);
        gemm_kernel<<<dim3(49, 12), 256, 0, stream>>>(
            hsh, wqkv + (size_t)d * 1536 * 512, qkv_b + d * 1536,
            nullptr, nullptr, nullptr, qb, kb, vb, 512, 1536, 0);
        attn_kernel<<<B_ * NH_ * 25, 128, 0, stream>>>(qb, kb, vb, oh);
        inv_shift_kernel<<<ELEM_BLKS, 256, 0, stream>>>(oh, osh);
        gemm_kernel<<<dim3(49, 4), 256, 0, stream>>>(
            osh, wprj + (size_t)d * 512 * 512, proj_b + d * C_,
            xp, y, nullptr, nullptr, nullptr, nullptr, 512, 512, 1);
        ln_kernel<<<M_ / 4, 128, 0, stream>>>(y, n2w + d * C_, n2b + d * C_, mbf, 0);
        gemm_kernel<<<dim3(49, 16), 256, 0, stream>>>(
            mbf, wfc1 + (size_t)d * 2048 * 512, fc1_b + d * 2048,
            nullptr, nullptr, hid, nullptr, nullptr, nullptr, 512, 2048, 2);
        gemm_kernel<<<dim3(49, 4), 256, 0, stream>>>(
            hid, wfc2 + (size_t)d * 512 * 2048, fc2_b + d * C_,
            y, xp, nullptr, nullptr, nullptr, nullptr, 2048, 512, 1);
    }
    transpose_out_kernel<<<ELEM_BLKS, 256, 0, stream>>>(xp, (float*)d_out);
}
